// Attention_26792005992653
// MI455X (gfx1250) — matrix operand order repaired and checked by reading, not compiled, not run on hardware
//
#include <hip/hip_runtime.h>

typedef __bf16 bhalf;
typedef bhalf v16bf __attribute__((ext_vector_type(16)));
typedef bhalf v8bf  __attribute__((ext_vector_type(8)));
typedef float v8f   __attribute__((ext_vector_type(8)));

#define S_   2048
#define DIN  1024
#define DOUT 128
#define NH   16
#define DQK  128
#define NQKV 6144   /* NH*(2*DQK+DOUT) */
#define HDIM 2048   /* NH*DOUT */

// ---- WMMA helpers -----------------------------------------------------------

static __device__ __forceinline__ v8f wmma_bf16(v16bf a, v16bf b, v8f c) {
  // D = A(16x32 bf16) x B(32x16 bf16) + C(16x16 f32)
  return __builtin_amdgcn_wmma_f32_16x16x32_bf16(false, a, false, b, (short)0, c,
                                                 false, false);
}

// A fragment (16x32, bf16): lane row m = lane%16; elements e<8 -> k=8*half+e,
// e>=8 -> k=16+8*half+(e-8)  ==> two contiguous 16B segments per lane.
static __device__ __forceinline__ v16bf loadA16x32(const bhalf* rowptr, int half) {
  v8bf lo = *(const v8bf*)(rowptr + 8 * half);
  v8bf hi = *(const v8bf*)(rowptr + 16 + 8 * half);
  return __builtin_shufflevector(lo, hi, 0, 1, 2, 3, 4, 5, 6, 7,
                                 8, 9, 10, 11, 12, 13, 14, 15);
}

// ---- f32 -> bf16 convert ----------------------------------------------------

__global__ void k_cvt(const float* __restrict__ in, bhalf* __restrict__ out, int n) {
  int i = blockIdx.x * blockDim.x + threadIdx.x;
  int stride = gridDim.x * blockDim.x;
  for (; i < n; i += stride) out[i] = (bhalf)in[i];
}

// ---- QKV projection GEMM: [4096,1024] x [1024,6144], scatter epilogue -------
// grid (32, 48), 256 threads (8 waves); wave tile = 16(M) x 128(N)

__global__ __launch_bounds__(256) void k_qkv(const bhalf* __restrict__ xb,
                                             const bhalf* __restrict__ wi,
                                             const float* __restrict__ vbias,
                                             bhalf* __restrict__ wq,
                                             bhalf* __restrict__ wkT,
                                             bhalf* __restrict__ wv) {
  const int wave = threadIdx.x >> 5, lane = threadIdx.x & 31;
  const int half = lane >> 4, l16 = lane & 15;
  const int m0 = blockIdx.x * 128 + wave * 16;
  const int n0 = blockIdx.y * 128;

  v8f acc[8] = {};
  const bhalf* arow = xb + (size_t)(m0 + l16) * DIN;
  for (int k0 = 0; k0 < DIN; k0 += 32) {
    v16bf a = loadA16x32(arow + k0, half);
    const bhalf* bp = wi + (size_t)k0 * NQKV + n0 + l16;
#pragma unroll
    for (int t = 0; t < 8; ++t) {
      v16bf b;
      for (int s = 0; s < 16; ++s) b[s] = bp[(size_t)((s & 7) + 8 * half + 16 * (s >> 3)) * NQKV + t * 16];
      acc[t] = wmma_bf16(a, b, acc[t]);
    }
  }
  // epilogue: C lane holds col n=l16, rows m=r+8*half. route q/k/v.
#pragma unroll
  for (int t = 0; t < 8; ++t) {
    int n = n0 + t * 16 + l16;
    int h = n / 384, c = n % 384;
#pragma unroll
    for (int r = 0; r < 8; ++r) {
      int gm = m0 + r + 8 * half;
      int z = gm >> 11, s = gm & (S_ - 1);
      float v = acc[t][r];
      if (c < 128) {
        wq[((size_t)(z * NH + h) * S_ + s) * DQK + c] = (bhalf)v;
      } else if (c < 256) {
        wkT[((size_t)(z * NH + h) * DQK + (c - 128)) * S_ + s] = (bhalf)v;
      } else {
        wv[((size_t)(z * NH + h) * S_ + s) * DOUT + (c - 256)] =
            (bhalf)(v + vbias[h * DOUT + (c - 256)]);
      }
    }
  }
}

// ---- RoPE (GPT-NeoX), in place on wq [zh][s][d] and wkT [zh][d][s] ----------

__global__ void k_rope(bhalf* __restrict__ wq, bhalf* __restrict__ wkT) {
  int id = blockIdx.x * blockDim.x + threadIdx.x;  // 2*16*2048*64 threads
  int j = id & 63;
  int s = (id >> 6) & (S_ - 1);
  int zh = id >> 17;
  // freq = 10000^(-2j/128) = exp(-j*ln(10000)/64); freq[j+64] == freq[j]
  float ang = (float)s * __expf(-(float)j * 0.14391156831212787f);
  float sn, cs;
  __sincosf(ang, &sn, &cs);

  bhalf* q = wq + ((size_t)zh * S_ + s) * DQK;
  float a = (float)q[j], b = (float)q[j + 64];
  q[j] = (bhalf)(a * cs - b * sn);
  q[j + 64] = (bhalf)(b * cs + a * sn);

  bhalf* k = wkT + (size_t)zh * DQK * S_;
  size_t i0 = (size_t)j * S_ + s, i1 = (size_t)(j + 64) * S_ + s;
  float ka = (float)k[i0], kb = (float)k[i1];
  k[i0] = (bhalf)(ka * cs - kb * sn);
  k[i1] = (bhalf)(kb * cs + ka * sn);
}

// ---- Flash attention: grid 512 = (z*16+h)*16 + qtile, 8 waves ---------------
// wave owns 16 q rows x 128 (DOUT); online softmax over 32 key blocks of 64.

__global__ __launch_bounds__(256) void k_attn(const bhalf* __restrict__ wq,
                                              const bhalf* __restrict__ wkT,
                                              const bhalf* __restrict__ wv,
                                              const unsigned char* __restrict__ mask,
                                              bhalf* __restrict__ oc) {
  __shared__ __align__(16) bhalf pls[8][16][72];  // padded rows: 144B stride
  const int wave = threadIdx.x >> 5, lane = threadIdx.x & 31;
  const int half = lane >> 4, l16 = lane & 15;
  const int zh = blockIdx.x >> 4, qt = blockIdx.x & 15;
  const int z = zh >> 4, h = zh & 15;
  const int qr0 = qt * 128 + wave * 16;
  const unsigned char* mz = mask + (size_t)z * S_;

  v16bf qa[4];
  {
    const bhalf* qrow = wq + ((size_t)zh * S_ + qr0 + l16) * DQK;
#pragma unroll
    for (int kc = 0; kc < 4; ++kc) qa[kc] = loadA16x32(qrow + kc * 32, half);
  }
  v8f o[8] = {};
  float rowmax[8], rowsum[8];
#pragma unroll
  for (int r = 0; r < 8; ++r) { rowmax[r] = -1e30f; rowsum[r] = 0.f; }

  const bhalf* kbase = wkT + (size_t)zh * DQK * S_;
  const bhalf* vbase = wv + (size_t)zh * S_ * DOUT;
  bhalf(*pw)[72] = pls[wave];

  for (int kb = 0; kb < 32; ++kb) {
    const int ks0 = kb * 64;
    // scores S = Q K^T : 16 wmma
    v8f sacc[4] = {};
#pragma unroll
    for (int kc = 0; kc < 4; ++kc) {
      const bhalf* bp = kbase + (size_t)(kc * 32) * S_ + ks0 + l16;
#pragma unroll
      for (int t = 0; t < 4; ++t) {
        v16bf b;
        for (int s = 0; s < 16; ++s) b[s] = bp[(size_t)((s & 7) + 8 * half + 16 * (s >> 3)) * S_ + t * 16];
        sacc[t] = wmma_bf16(qa[kc], b, sacc[t]);
      }
    }
    // scale + key mask (lane owns one key column per tile)
#pragma unroll
    for (int t = 0; t < 4; ++t) {
      bool mk = mz[ks0 + t * 16 + l16] != 0;
#pragma unroll
      for (int r = 0; r < 8; ++r) {
        float f = sacc[t][r] * 0.08838834764831845f;  // 1/sqrt(128)
        sacc[t][r] = mk ? -1e30f : f;
      }
    }
    // per-row online max (reduce across the 16 lanes of this half)
    float scl[8], bm[8], bs[8];
#pragma unroll
    for (int r = 0; r < 8; ++r) {
      float m = fmaxf(fmaxf(sacc[0][r], sacc[1][r]), fmaxf(sacc[2][r], sacc[3][r]));
      m = fmaxf(m, __shfl_xor(m, 1));
      m = fmaxf(m, __shfl_xor(m, 2));
      m = fmaxf(m, __shfl_xor(m, 4));
      m = fmaxf(m, __shfl_xor(m, 8));
      float nm = fmaxf(rowmax[r], m);
      scl[r] = __expf(rowmax[r] - nm);
      rowmax[r] = nm;
      bm[r] = nm;
      bs[r] = 0.f;
    }
    // P = exp(S - max), partial row sums
#pragma unroll
    for (int t = 0; t < 4; ++t)
#pragma unroll
      for (int r = 0; r < 8; ++r) {
        float p = __expf(sacc[t][r] - bm[r]);
        sacc[t][r] = p;
        bs[r] += p;
      }
#pragma unroll
    for (int r = 0; r < 8; ++r) {
      float s = bs[r];
      s += __shfl_xor(s, 1);
      s += __shfl_xor(s, 2);
      s += __shfl_xor(s, 4);
      s += __shfl_xor(s, 8);
      rowsum[r] = rowsum[r] * scl[r] + s;
    }
    // rescale O accumulators
#pragma unroll
    for (int nt = 0; nt < 8; ++nt)
#pragma unroll
      for (int r = 0; r < 8; ++r) o[nt][r] *= scl[r];
    // P: C layout -> LDS -> A layout (wave-private region; DS is in-order)
#pragma unroll
    for (int t = 0; t < 4; ++t)
#pragma unroll
      for (int r = 0; r < 8; ++r)
        pw[r + 8 * half][t * 16 + l16] = (bhalf)sacc[t][r];
    // O += P V : 16 wmma
#pragma unroll
    for (int c2 = 0; c2 < 2; ++c2) {
      const bhalf* prow = &pw[l16][c2 * 32];
      v8bf lo = *(const v8bf*)(prow + 8 * half);
      v8bf hi = *(const v8bf*)(prow + 16 + 8 * half);
      v16bf pa = __builtin_shufflevector(lo, hi, 0, 1, 2, 3, 4, 5, 6, 7,
                                         8, 9, 10, 11, 12, 13, 14, 15);
      const bhalf* vp = vbase + (size_t)(ks0 + c2 * 32) * DOUT + l16;
#pragma unroll
      for (int nt = 0; nt < 8; ++nt) {
        v16bf vb;
        for (int s = 0; s < 16; ++s) vb[s] = vp[(size_t)((s & 7) + 8 * half + 16 * (s >> 3)) * DOUT + nt * 16];
        o[nt] = wmma_bf16(pa, vb, o[nt]);
      }
    }
  }
  // normalize, query-mask (nan_to_num), cube, store bf16 [z*s][h*128+x]
  float inv[8];
  int srow[8];
#pragma unroll
  for (int r = 0; r < 8; ++r) {
    int sq = qr0 + r + 8 * half;
    srow[r] = sq;
    bool qm = mz[sq] != 0;
    inv[r] = (qm || rowsum[r] <= 0.f) ? 0.f : 1.f / rowsum[r];
  }
#pragma unroll
  for (int nt = 0; nt < 8; ++nt)
#pragma unroll
    for (int r = 0; r < 8; ++r) {
      float v = o[nt][r] * inv[r];
      v = v * v * v;
      oc[(size_t)(z * S_ + srow[r]) * HDIM + h * DQK + nt * 16 + l16] = (bhalf)v;
    }
}

// ---- Output projection: [4096,2048] x [2048,128], bias + cube ---------------

__global__ __launch_bounds__(256) void k_out(const bhalf* __restrict__ oc,
                                             const bhalf* __restrict__ wo,
                                             const float* __restrict__ bias,
                                             float* __restrict__ out) {
  const int wave = threadIdx.x >> 5, lane = threadIdx.x & 31;
  const int half = lane >> 4, l16 = lane & 15;
  const int m0 = blockIdx.x * 128 + wave * 16;

  v8f acc[8] = {};
  const bhalf* arow = oc + (size_t)(m0 + l16) * HDIM;
  for (int k0 = 0; k0 < HDIM; k0 += 32) {
    v16bf a = loadA16x32(arow + k0, half);
    const bhalf* bp = wo + (size_t)k0 * DOUT + l16;
#pragma unroll
    for (int t = 0; t < 8; ++t) {
      v16bf b;
      for (int s = 0; s < 16; ++s) b[s] = bp[(size_t)((s & 7) + 8 * half + 16 * (s >> 3)) * DOUT + t * 16];
      acc[t] = wmma_bf16(a, b, acc[t]);
    }
  }
#pragma unroll
  for (int t = 0; t < 8; ++t) {
    int y = t * 16 + l16;
    float bb = bias[y];
#pragma unroll
    for (int r = 0; r < 8; ++r) {
      int gm = m0 + r + 8 * half;
      float v = acc[t][r] + bb;
      v = v * v * v;
      out[(size_t)gm * DOUT + y] = v;
    }
  }
}

// ---- launcher ---------------------------------------------------------------

extern "C" void kernel_launch(void* const* d_in, const int* in_sizes, int n_in,
                              void* d_out, int out_size, void* d_ws, size_t ws_size,
                              hipStream_t stream) {
  const float* x = (const float*)d_in[0];
  const unsigned char* mask = (const unsigned char*)d_in[1];  // jnp bool_
  const float* proj_in = (const float*)d_in[2];
  const float* v_bias = (const float*)d_in[3];
  const float* proj_out = (const float*)d_in[4];
  const float* proj_out_bias = (const float*)d_in[5];
  float* out = (float*)d_out;

  char* ws = (char*)d_ws;
  bhalf* xb  = (bhalf*)(ws);              // [4096][1024]           8 MiB
  bhalf* wi  = (bhalf*)(ws + 8388608);    // [1024][6144]          12 MiB
  bhalf* wq  = (bhalf*)(ws + 20971520);   // [2*16][2048][128]     16 MiB
  bhalf* wkT = (bhalf*)(ws + 37748736);   // [2*16][128][2048]     16 MiB
  bhalf* wv  = (bhalf*)(ws + 54525952);   // [2*16][2048][128]     16 MiB
  bhalf* oc  = (bhalf*)(ws + 71303168);   // [4096][2048]          16 MiB
  bhalf* wo  = (bhalf*)(ws + 88080384);   // [2048][128]          0.5 MiB

  k_cvt<<<2048, 256, 0, stream>>>(x, xb, 4096 * 1024);
  k_cvt<<<3072, 256, 0, stream>>>(proj_in, wi, 1024 * 6144);
  k_cvt<<<128, 256, 0, stream>>>(proj_out, wo, 2048 * 128);
  k_qkv<<<dim3(32, 48), 256, 0, stream>>>(xb, wi, v_bias, wq, wkT, wv);
  k_rope<<<16384, 256, 0, stream>>>(wq, wkT);
  k_attn<<<512, 256, 0, stream>>>(wq, wkT, wv, mask, oc);
  k_out<<<32, 256, 0, stream>>>(oc, wo, proj_out_bias, out);
}
